// TransformerBlock_3015067041875
// MI455X (gfx1250) — compile-verified, matrix operand order checked by reading, not run on hardware
//
#include <hip/hip_runtime.h>
#include <hip/hip_bf16.h>
#include <math.h>

typedef __attribute__((ext_vector_type(16))) _Float16 v16h;
typedef __attribute__((ext_vector_type(8)))  float    v8f;
typedef unsigned int u32x4 __attribute__((ext_vector_type(4)));
typedef int          i32x8 __attribute__((ext_vector_type(8)));
typedef int          i32x4 __attribute__((ext_vector_type(4)));

#define D_MODEL   1024
#define N_HEADS   16
#define HEAD_DIM  64
#define D_FF      4096
#define SEQ_T     2048
#define BATCH     2
#define M_ROWS    (SEQ_T * BATCH)   // 4096

// ---------------------------------------------------------------------------
// Fragment index helper: ISA 7.12.2, 16-bit A-matrix 16x32.
// VGPR v holds K pair starting at: (v<4 ? 2v : 16+2(v-4)) + 8*(lane>=16)
// ---------------------------------------------------------------------------
__device__ __forceinline__ int frag_k(int v, int half) {
    return (v < 4 ? 2 * v : 16 + 2 * (v - 4)) + half * 8;
}

// ===========================================================================
// GEMM: C[M,N] = epilogue( A[M,K] @ W[K,N] + bias )   (W row-major [K,N])
// ACT: 0 = none, 1 = SiLU.  HAS_RES: C += res.
// Block: 256 threads (8 waves). Tile: BM=128, BN=64, BK=32.
// A tile staged manually (fp32->f16). B (weight) tile fetched by the Tensor
// Data Mover: one wave issues TENSOR_LOAD_TO_LDS with a 2-D D# descriptor
// (ISA ch.8); completion via s_wait_tensorcnt + workgroup barrier.
// Waves 4x2; each wave computes 32x32 via 2x2 v_wmma_f32_16x16x32_f16.
// ===========================================================================
template <int ACT, bool HAS_RES>
__global__ __launch_bounds__(256) void gemm_wmma(
    const float* __restrict__ A, const float* __restrict__ W,
    const float* __restrict__ bias, const float* __restrict__ res,
    float* __restrict__ C, int M, int N, int K)
{
    constexpr int BM = 128, BN = 64, BK = 32;
    __shared__ _Float16 Alds[BM][BK + 8];
    __shared__ float    Bf32[BK][BN];      // TDM destination (packed fp32 tile)

    const int tid  = threadIdx.x;
    const int lane = tid & 31;
    const int wave = tid >> 5;
    const int half = lane >> 4;
    const int lr   = lane & 15;
    const int wm   = wave & 3;     // 4 waves along M
    const int wn   = wave >> 2;    // 2 waves along N
    const int m_block = blockIdx.x * BM;
    const int n_block = blockIdx.y * BN;
    const int wrow = m_block + wm * 32;
    const int wcol = n_block + wn * 32;

    // flat shared pointer low dword == workgroup-relative LDS byte offset
    const unsigned lds_b_off = (unsigned)(uintptr_t)(&Bf32[0][0]);

    const v8f vzero = {0.f, 0.f, 0.f, 0.f, 0.f, 0.f, 0.f, 0.f};
    v8f acc[2][2];
    for (int mi = 0; mi < 2; ++mi)
        for (int ni = 0; ni < 2; ++ni) acc[mi][ni] = vzero;

    for (int k0 = 0; k0 < K; k0 += BK) {
        __syncthreads();   // previous tile fully consumed

        // --- TDM: B tile 32 rows x 64 cols of fp32 from W[k0.., n_block..]
        if (wave == 0) {
            const unsigned long long gaddr =
                (unsigned long long)(uintptr_t)(W + (size_t)k0 * N + n_block);
            u32x4 g0;
            g0[0] = 1u;                                   // count=1 (valid D#)
            g0[1] = lds_b_off;                            // lds_addr (bytes)
            g0[2] = (unsigned)(gaddr & 0xFFFFFFFFull);    // global_addr[31:0]
            g0[3] = (unsigned)((gaddr >> 32) & 0x1FFFFFFull) | (2u << 30); // [56:32] | type=2
            i32x8 g1;
            g1[0] = (int)(2u << 16);                      // data_size=4B; mask=0
            g1[1] = (int)(((unsigned)N & 0xFFFFu) << 16); // tensor_dim0 lo16
            g1[2] = (int)((((unsigned)N >> 16) & 0xFFFFu) | (32u << 16)); // td0 hi | td1 lo (=32 rows)
            g1[3] = (int)(64u << 16);                     // td1 hi=0 | tile_dim0=64
            g1[4] = (int)32u;                             // tile_dim1=32 | tile_dim2=0
            g1[5] = (int)N;                               // tensor_dim0_stride lo32 (elements)
            g1[6] = 0;                                    // stride0 hi | stride1 lo
            g1[7] = 0;                                    // stride1 hi
            const i32x4 gz4 = {0, 0, 0, 0};               // 2-D: groups 2/3 unused
            const i32x8 gz8 = {0, 0, 0, 0, 0, 0, 0, 0};
            __builtin_amdgcn_tensor_load_to_lds(g0, g1, gz4, gz4, gz8, 0);
        }

        // --- stage A tile 128x32 (fp32 -> f16), float4 loads, 16 elems/thread
        {
            const int kk = (tid & 7) * 4;   // 0..28
            const int r0 = tid >> 3;        // 0..31
            for (int i = 0; i < 4; ++i) {
                const int r = r0 + i * 32;
                const float4 v = *(const float4*)(A + (size_t)(m_block + r) * K + k0 + kk);
                Alds[r][kk + 0] = (_Float16)v.x;
                Alds[r][kk + 1] = (_Float16)v.y;
                Alds[r][kk + 2] = (_Float16)v.z;
                Alds[r][kk + 3] = (_Float16)v.w;
            }
        }
        if (k0 + BK < K) {  // hint next A tile toward L2 (global_prefetch_b8)
            __builtin_prefetch(A + (size_t)(m_block + (tid >> 3)) * K + k0 + BK, 0, 1);
        }
        __builtin_amdgcn_s_wait_tensorcnt(0);   // TDM tile landed (wave 0)
        __syncthreads();

        // --- build fragments (ISA 7.12.2 layouts) and multiply
        v16h af[2], bf[2];
        for (int mi = 0; mi < 2; ++mi) {
            const int row = wm * 32 + mi * 16 + lr;
            for (int v = 0; v < 8; ++v) {
                const int k = frag_k(v, half);
                af[mi][2 * v + 0] = Alds[row][k];
                af[mi][2 * v + 1] = Alds[row][k + 1];
            }
        }
        for (int ni = 0; ni < 2; ++ni) {
            const int col = wn * 32 + ni * 16 + lr;
            for (int v = 0; v < 8; ++v) {
                const int k = frag_k(v, half);
                bf[ni][2 * v + 0] = (_Float16)Bf32[k][col];
                bf[ni][2 * v + 1] = (_Float16)Bf32[k + 1][col];
            }
        }
        for (int mi = 0; mi < 2; ++mi)
            for (int ni = 0; ni < 2; ++ni)
                acc[mi][ni] = __builtin_amdgcn_wmma_f32_16x16x32_f16(
                    false, af[mi], false, bf[ni], (short)0, acc[mi][ni], false, false);
    }

    // --- epilogue: C/D layout (VGPR v <-> row v + 8*half, col = lr)
    for (int mi = 0; mi < 2; ++mi)
        for (int ni = 0; ni < 2; ++ni)
            for (int v = 0; v < 8; ++v) {
                const int row = wrow + mi * 16 + half * 8 + v;
                const int col = wcol + ni * 16 + lr;
                float val = acc[mi][ni][v] + bias[col];
                if (ACT == 1) val = val * (1.0f / (1.0f + __expf(-val)));  // SiLU
                if (HAS_RES) val += res[(size_t)row * N + col];
                C[(size_t)row * N + col] = val;
            }
}

// ===========================================================================
// Flash attention, causal, H=16, hd=64.
// Block: 256 threads = 8 waves; 128 query rows per block, one (b,h) per
// blockIdx.y. Key blocks of 32 staged in LDS; online softmax in registers.
// ===========================================================================
__global__ __launch_bounds__(256) void attn_flash(
    const float* __restrict__ qkv, float* __restrict__ ctx)
{
    __shared__ _Float16 Qlds[128][HEAD_DIM + 8];
    __shared__ _Float16 Klds[32][HEAD_DIM + 8];
    __shared__ _Float16 Vlds[32][HEAD_DIM + 8];
    __shared__ _Float16 Plds[8][16][40];   // per-wave P bounce buffer

    const int tid  = threadIdx.x;
    const int lane = tid & 31;
    const int wave = tid >> 5;
    const int half = lane >> 4;
    const int lr   = lane & 15;
    const int bh   = blockIdx.y;
    const int b    = bh / N_HEADS;
    const int hh   = bh % N_HEADS;
    const int q_base = blockIdx.x * 128;
    const size_t stride3 = 3 * D_MODEL;

    // --- stage Q tile 128x64 (f16)
    {
        const int cc = (tid & 15) * 4;
        const int r0 = tid >> 4;
        for (int i = 0; i < 8; ++i) {
            const int r = r0 + i * 16;
            const int t = q_base + r;
            const float4 v = *(const float4*)(qkv + ((size_t)t * BATCH + b) * stride3 +
                                              hh * HEAD_DIM + cc);
            Qlds[r][cc + 0] = (_Float16)v.x;
            Qlds[r][cc + 1] = (_Float16)v.y;
            Qlds[r][cc + 2] = (_Float16)v.z;
            Qlds[r][cc + 3] = (_Float16)v.w;
        }
    }
    __syncthreads();

    // --- Q fragments: two K=32 chunks of hd=64, kept live across the loop
    v16h qf[2];
    {
        const int qrow = wave * 16 + lr;
        for (int c = 0; c < 2; ++c)
            for (int v = 0; v < 8; ++v) {
                const int k = c * 32 + frag_k(v, half);
                qf[c][2 * v + 0] = Qlds[qrow][k];
                qf[c][2 * v + 1] = Qlds[qrow][k + 1];
            }
    }

    const v8f vzero = {0.f, 0.f, 0.f, 0.f, 0.f, 0.f, 0.f, 0.f};
    v8f o[4];
    for (int nb = 0; nb < 4; ++nb) o[nb] = vzero;
    float m_run[8], l_run[8];
    for (int v = 0; v < 8; ++v) { m_run[v] = -1e30f; l_run[v] = 0.f; }

    const int q_wave_base = q_base + wave * 16;
    const int nkb = (q_base + 128) / 32;      // causal extent for this block
    const float scale = 0.125f;               // 1/sqrt(64)

    for (int kb = 0; kb < nkb; ++kb) {
        __syncthreads();
        // --- stage K,V tiles 32x64 each
        {
            const int cc = (tid & 15) * 4;
            const int r0 = tid >> 4;
            for (int i = 0; i < 2; ++i) {
                const int r = r0 + i * 16;
                const int t = kb * 32 + r;
                const size_t base = ((size_t)t * BATCH + b) * stride3 + hh * HEAD_DIM + cc;
                const float4 kv = *(const float4*)(qkv + base + D_MODEL);
                const float4 vv = *(const float4*)(qkv + base + 2 * D_MODEL);
                Klds[r][cc + 0] = (_Float16)kv.x; Klds[r][cc + 1] = (_Float16)kv.y;
                Klds[r][cc + 2] = (_Float16)kv.z; Klds[r][cc + 3] = (_Float16)kv.w;
                Vlds[r][cc + 0] = (_Float16)vv.x; Vlds[r][cc + 1] = (_Float16)vv.y;
                Vlds[r][cc + 2] = (_Float16)vv.z; Vlds[r][cc + 3] = (_Float16)vv.w;
            }
        }
        __syncthreads();
        if (kb * 32 > q_wave_base + 15) continue;   // whole key block above diagonal

        // --- S = (Q @ K^T) * scale, two 16x16 key blocks, hd split K=32+32
        v8f s[2];
        for (int nb = 0; nb < 2; ++nb) {
            s[nb] = vzero;
            for (int c = 0; c < 2; ++c) {
                v16h kf;  // B-matrix fragment of K^T: element (k,n) = K[key=n][hd=c*32+k]
                for (int v = 0; v < 8; ++v) {
                    const int k = frag_k(v, half);
                    kf[2 * v + 0] = Klds[nb * 16 + lr][c * 32 + k];
                    kf[2 * v + 1] = Klds[nb * 16 + lr][c * 32 + k + 1];
                }
                s[nb] = __builtin_amdgcn_wmma_f32_16x16x32_f16(
                    false, qf[c], false, kf, (short)0, s[nb], false, false);
            }
        }

        // --- mask + online softmax (row stats per lane-half via shfl_xor)
        float rmax[8];
        for (int v = 0; v < 8; ++v) {
            const int qg = q_wave_base + half * 8 + v;
            for (int nb = 0; nb < 2; ++nb) {
                const int kg = kb * 32 + nb * 16 + lr;
                float x = s[nb][v] * scale;
                if (kg > qg) x = -1e30f;
                s[nb][v] = x;
            }
            rmax[v] = fmaxf(s[0][v], s[1][v]);
        }
        for (int off = 1; off < 16; off <<= 1)
            for (int v = 0; v < 8; ++v)
                rmax[v] = fmaxf(rmax[v], __shfl_xor(rmax[v], off, 32));

        float alpha[8], rsum[8];
        for (int v = 0; v < 8; ++v) {
            const float m_new = fmaxf(m_run[v], rmax[v]);
            alpha[v] = __expf(m_run[v] - m_new);
            m_run[v] = m_new;
            const float p0 = __expf(s[0][v] - m_new);
            const float p1 = __expf(s[1][v] - m_new);
            s[0][v] = p0; s[1][v] = p1;
            rsum[v] = p0 + p1;
        }
        for (int off = 1; off < 16; off <<= 1)
            for (int v = 0; v < 8; ++v)
                rsum[v] += __shfl_xor(rsum[v], off, 32);
        for (int v = 0; v < 8; ++v) l_run[v] = l_run[v] * alpha[v] + rsum[v];
        for (int nb = 0; nb < 4; ++nb)
            for (int v = 0; v < 8; ++v) o[nb][v] *= alpha[v];

        // --- P: C-layout -> A-fragment layout via per-wave LDS bounce
        for (int nb = 0; nb < 2; ++nb)
            for (int v = 0; v < 8; ++v)
                Plds[wave][half * 8 + v][nb * 16 + lr] = (_Float16)s[nb][v];
        asm volatile("s_wait_dscnt 0" ::: "memory");  // CDNA5 split DS counter

        v16h pf;
        for (int v = 0; v < 8; ++v) {
            const int k = frag_k(v, half);
            pf[2 * v + 0] = Plds[wave][lr][k];
            pf[2 * v + 1] = Plds[wave][lr][k + 1];
        }

        // --- O += P @ V  (4 column blocks of hd)
        for (int nb = 0; nb < 4; ++nb) {
            v16h vf;  // B fragment: element (k,n) = V[key=k][hd = nb*16+n]
            for (int v = 0; v < 8; ++v) {
                const int k = frag_k(v, half);
                vf[2 * v + 0] = Vlds[k][nb * 16 + lr];
                vf[2 * v + 1] = Vlds[k + 1][nb * 16 + lr];
            }
            o[nb] = __builtin_amdgcn_wmma_f32_16x16x32_f16(
                false, pf, false, vf, (short)0, o[nb], false, false);
        }
    }

    // --- normalize and write ctx [T*B, D]
    for (int nb = 0; nb < 4; ++nb)
        for (int v = 0; v < 8; ++v) {
            const int t   = q_wave_base + half * 8 + v;
            const int col = hh * HEAD_DIM + nb * 16 + lr;
            const float inv = 1.0f / l_run[v];
            ctx[((size_t)t * BATCH + b) * D_MODEL + col] = o[nb][v] * inv;
        }
}

// ===========================================================================
// RMSNorm: one row (1024 f32) per block of 256 threads.
// ===========================================================================
__global__ __launch_bounds__(256) void rmsnorm_kernel(
    const float* __restrict__ X, const float* __restrict__ gain,
    float* __restrict__ out)
{
    __shared__ float red[8];
    const int row = blockIdx.x;
    const int tid = threadIdx.x;
    const float* xr = X + (size_t)row * D_MODEL;

    float vals[4];
    float ss = 0.f;
    for (int i = 0; i < 4; ++i) {
        vals[i] = xr[tid + i * 256];
        ss += vals[i] * vals[i];
    }
    for (int off = 1; off < 32; off <<= 1) ss += __shfl_xor(ss, off, 32);
    if ((tid & 31) == 0) red[tid >> 5] = ss;
    __syncthreads();
    float tot = 0.f;
    for (int i = 0; i < 8; ++i) tot += red[i];
    const float rms = rsqrtf(tot * (1.0f / D_MODEL) + 1e-5f);
    for (int i = 0; i < 4; ++i) {
        const int col = tid + i * 256;
        out[(size_t)row * D_MODEL + col] = vals[i] * rms * gain[col];
    }
}

// ===========================================================================
extern "C" void kernel_launch(void* const* d_in, const int* in_sizes, int n_in,
                              void* d_out, int out_size, void* d_ws, size_t ws_size,
                              hipStream_t stream)
{
    (void)in_sizes; (void)n_in; (void)out_size; (void)ws_size;
    const float* x      = (const float*)d_in[0];
    const float* w_qkv  = (const float*)d_in[1];
    const float* b_qkv  = (const float*)d_in[2];
    const float* w_out  = (const float*)d_in[3];
    const float* b_out  = (const float*)d_in[4];
    const float* w1     = (const float*)d_in[5];
    const float* b1     = (const float*)d_in[6];
    const float* w2     = (const float*)d_in[7];
    const float* b2     = (const float*)d_in[8];
    const float* gain   = (const float*)d_in[9];

    float* ws   = (float*)d_ws;
    float* qkv  = ws;                                   // 4096 x 3072
    float* ctx  = qkv + (size_t)M_ROWS * 3 * D_MODEL;   // 4096 x 1024
    float* x1   = ctx + (size_t)M_ROWS * D_MODEL;       // 4096 x 1024
    float* hbuf = x1  + (size_t)M_ROWS * D_MODEL;       // 4096 x 4096
    float* x2   = hbuf + (size_t)M_ROWS * D_FF;         // 4096 x 1024

    const dim3 blk(256);

    // 1) qkv = x @ w_qkv + b_qkv
    gemm_wmma<0, false><<<dim3(M_ROWS / 128, (3 * D_MODEL) / 64), blk, 0, stream>>>(
        x, w_qkv, b_qkv, nullptr, qkv, M_ROWS, 3 * D_MODEL, D_MODEL);

    // 2) ctx = causal_softmax(QK^T/8) @ V   (flash, per (b,h))
    attn_flash<<<dim3(SEQ_T / 128, BATCH * N_HEADS), blk, 0, stream>>>(qkv, ctx);

    // 3) x1 = x + ctx @ w_out + b_out
    gemm_wmma<0, true><<<dim3(M_ROWS / 128, D_MODEL / 64), blk, 0, stream>>>(
        ctx, w_out, b_out, x, x1, M_ROWS, D_MODEL, D_MODEL);

    // 4) h = silu(x1 @ w1 + b1)
    gemm_wmma<1, false><<<dim3(M_ROWS / 128, D_FF / 64), blk, 0, stream>>>(
        x1, w1, b1, nullptr, hbuf, M_ROWS, D_FF, D_MODEL);

    // 5) x2 = x1 + h @ w2 + b2
    gemm_wmma<0, true><<<dim3(M_ROWS / 128, D_MODEL / 64), blk, 0, stream>>>(
        hbuf, w2, b2, x1, x2, M_ROWS, D_MODEL, D_FF);

    // 6) out = rmsnorm(x2) * gain
    rmsnorm_kernel<<<dim3(M_ROWS), blk, 0, stream>>>(x2, gain, (float*)d_out);
}
